// SparseSelfAttention_43490838839766
// MI455X (gfx1250) — matrix operand order repaired and checked by reading, not compiled, not run on hardware
//
#include <hip/hip_runtime.h>
#include <cstdint>

// ---------------------------------------------------------------------------
// SparseSelfAttention for MI455X (gfx1250, wave32, WMMA bf16 + TDM async path)
// B=4, S=2048, D=1024, H=16, hd=64, WIN=64, BLOCK=16, nw=32
// ---------------------------------------------------------------------------

typedef __attribute__((ext_vector_type(16))) __bf16        bf16x16;
typedef __attribute__((ext_vector_type(8)))  float         f32x8;
typedef __attribute__((ext_vector_type(4)))  unsigned int  u32x4;
typedef __attribute__((ext_vector_type(8)))  int           i32x8;
typedef __attribute__((ext_vector_type(4)))  int           i32x4;

union BF16x16 {
    bf16x16 v;
    __bf16  e[16];
    uint4   q[2];
};

#define WMMA_BF16(A, B, C) \
    __builtin_amdgcn_wmma_f32_16x16x32_bf16(false, (A), false, (B), (short)0, (C), false, false)

static constexpr int   Bsz  = 4;
static constexpr int   Ssz  = 2048;
static constexpr int   Dsz  = 1024;
static constexpr int   Hsz  = 16;
static constexpr int   HD   = 64;
static constexpr int   WIN  = 64;
static constexpr int   NW   = 32;          // S / WIN
static constexpr int   Mtot = Bsz * Ssz;   // 8192
static constexpr float NEGV = -1000000000.0f;

// ---------------------------------------------------------------------------
// Tensor Data Mover: async-load a 2D tile (tile_k elems contiguous per row,
// tile_n rows, row stride = stride_elems) of bf16 data into LDS at lds_off.
// D# layout per CDNA5 ISA.  Issue once per wave.
// This toolchain's builtin is the 6-arg variant:
//   (u32x4 g0, i32x8 g1, i32x4 g2, i32x4 g3, i32x8 extra, i32 cpol)
// ---------------------------------------------------------------------------
__device__ __forceinline__ void tdm_load_tile_bf16(const __bf16* gptr,
                                                   unsigned lds_off,
                                                   unsigned tile_k, unsigned tile_n,
                                                   unsigned tensor_d0, unsigned tensor_d1,
                                                   unsigned stride_elems) {
    const unsigned long long ga = (unsigned long long)(uintptr_t)gptr;
    u32x4 g0;
    g0[0] = 1u;                                            // count=1, user-mode, no gather
    g0[1] = lds_off;                                       // lds_addr (bytes)
    g0[2] = (unsigned)(ga & 0xffffffffull);                // global_addr[31:0]
    g0[3] = (unsigned)((ga >> 32) & 0x01ffffffull)         // global_addr[56:32]
          | (2u << 30);                                    // type = 2 ("image")
    i32x8 g1;
    g1[0] = 0x00010000;                                    // wg_mask=0, data_size=1 (2B)
    g1[1] = (int)((tensor_d0 & 0xffffu) << 16);            // tensor_dim0[15:0]
    g1[2] = (int)(((tensor_d0 >> 16) & 0xffffu)            // tensor_dim0[31:16]
          | ((tensor_d1 & 0xffffu) << 16));                // tensor_dim1[15:0]
    g1[3] = (int)(((tensor_d1 >> 16) & 0xffffu)            // tensor_dim1[31:16]
          | ((tile_k & 0xffffu) << 16));                   // tile_dim0
    g1[4] = (int)(tile_n & 0xffffu);                       // tile_dim1 (tile_dim2=0)
    g1[5] = (int)stride_elems;                             // tensor_dim0_stride[31:0]
    g1[6] = 0;                                             // stride hi / dim1_stride
    g1[7] = 0;
    i32x4 z4 = {};
    i32x8 z8 = {};
    __builtin_amdgcn_tensor_load_to_lds(g0, g1, z4, z4, z8, 0);
}

// ---------------------------------------------------------------------------
// Kernel 1: convert x (fp32) -> bf16, row-major (8192 x 1024)
// ---------------------------------------------------------------------------
__global__ void cvt_x_kernel(const float* __restrict__ x, __bf16* __restrict__ xb, int n) {
    int i = blockIdx.x * blockDim.x + threadIdx.x;
    if (i < n) xb[i] = (__bf16)x[i];
}

// ---------------------------------------------------------------------------
// Kernel 2: convert + transpose Wq/Wk/Wv (fp32, DxD) -> Wt bf16 (Wt[n][k] = W[k][n])
// so that WMMA B-fragments are contiguous 8-element runs.
// ---------------------------------------------------------------------------
__global__ void cvt_wt_kernel(const float* __restrict__ Wq, const float* __restrict__ Wk,
                              const float* __restrict__ Wv, __bf16* __restrict__ wt) {
    int z = blockIdx.z;
    const float* W = (z == 0) ? Wq : (z == 1) ? Wk : Wv;
    int i = blockIdx.x * blockDim.x + threadIdx.x;   // 0 .. D*D-1
    int k = i >> 10;
    int n = i & (Dsz - 1);
    wt[(size_t)z * Dsz * Dsz + (size_t)n * Dsz + k] = (__bf16)W[(size_t)k * Dsz + n];
}

// ---------------------------------------------------------------------------
// Kernel 3: projection GEMM  Y = Xbf16 @ W + b,  stored as (B,H,S,hd) bf16.
// grid = (Mtot/128, D/64, 3); block = 256 (8 waves). Each wave: 16x64 tile.
// Weight tile (64 cols x 32 k = 4KB) is shared by all 8 waves: staged in LDS
// by the Tensor Data Mover, double-buffered so TDM(k+32) overlaps WMMA(k).
// z=0 -> Q (pre-scaled by 1/sqrt(hd)=0.125), z=1 -> K, z=2 -> V.
// ---------------------------------------------------------------------------
__global__ __launch_bounds__(256) void proj_kernel(const __bf16* __restrict__ xb,
                                                   const __bf16* __restrict__ wt,
                                                   const float* __restrict__ bq,
                                                   const float* __restrict__ bk,
                                                   const float* __restrict__ bv,
                                                   __bf16* __restrict__ qkv) {
    __shared__ __align__(128) __bf16 btile[2][64 * 32];   // 2 x 4 KB double buffer

    const int z          = blockIdx.z;
    const __bf16* Wt     = wt + (size_t)z * Dsz * Dsz;
    const float*  bias   = (z == 0) ? bq : (z == 1) ? bk : bv;
    __bf16*       out    = qkv + (size_t)z * Mtot * Dsz;
    const float   scale  = (z == 0) ? 0.125f : 1.0f;

    const int wave = threadIdx.x >> 5;
    const int lane = threadIdx.x & 31;
    const int hi   = lane >> 4;      // lane group (0: lanes 0-15, 1: lanes 16-31)
    const int nn   = lane & 15;

    const int m0 = blockIdx.x * 128 + wave * 16;
    const int n0 = blockIdx.y * 64;
    const int arow = m0 + nn;        // A-matrix row for this lane

    const unsigned lds0 = (unsigned)(uintptr_t)&btile[0][0];
    const unsigned lds1 = (unsigned)(uintptr_t)&btile[1][0];

    // Prologue: TDM tile for k=0 into buffer 0 (wave 0 issues; EXEC ignored by TDM)
    if (wave == 0) {
        tdm_load_tile_bf16(Wt + (size_t)n0 * Dsz, lds0, 32, 64, Dsz, Dsz, Dsz);
        __builtin_amdgcn_s_wait_tensorcnt(0);
    }
    __syncthreads();

    f32x8 acc[4] = {{}, {}, {}, {}};
    int buf = 0;

    for (int k = 0; k < Dsz; k += 32) {
        // Kick off async TDM for the next weight tile into the other buffer
        if (wave == 0 && (k + 32) < Dsz) {
            tdm_load_tile_bf16(Wt + (size_t)n0 * Dsz + (k + 32),
                               buf ? lds0 : lds1, 32, 64, Dsz, Dsz, Dsz);
        }

        // A fragment (16x32 bf16) direct from global:
        // lane<16 -> K {0..7,16..23}; lane>=16 -> K {8..15,24..31}
        BF16x16 a;
        const __bf16* ap = xb + (size_t)arow * Dsz + k + hi * 8;
        a.q[0] = *(const uint4*)(ap);
        a.q[1] = *(const uint4*)(ap + 16);

        // B fragments (32x16 bf16) from the LDS-staged weight tile:
        // lane n = column; lane<16 -> K {0..7,16..23}; lane>=16 -> K {8..15,24..31}
        BF16x16 bf[4];
#pragma unroll
        for (int f = 0; f < 4; ++f) {
            const __bf16* bp = &btile[buf][(f * 16 + nn) * 32 + hi * 8];
            bf[f].q[0] = *(const uint4*)(bp);
            bf[f].q[1] = *(const uint4*)(bp + 16);
        }
#pragma unroll
        for (int f = 0; f < 4; ++f)
            acc[f] = WMMA_BF16(a.v, bf[f].v, acc[f]);

        // Next tile must be resident and everyone done reading this one
        if (wave == 0) __builtin_amdgcn_s_wait_tensorcnt(0);
        __syncthreads();
        buf ^= 1;
    }

    // Epilogue: add bias, scale (Q only), convert to bf16, store (B,H,S,hd)
#pragma unroll
    for (int f = 0; f < 4; ++f) {
        const int   n  = n0 + f * 16 + nn;
        const float bb = bias[n];
        const int   h  = n >> 6;
        const int   d  = n & 63;
#pragma unroll
        for (int r = 0; r < 8; ++r) {
            const int m = m0 + r + hi * 8;          // C/D: M = r + 8*hi
            const int bidx = m >> 11;               // / 2048
            const int s    = m & 2047;
            out[(((size_t)bidx * Hsz + h) * Ssz + s) * HD + d] =
                (__bf16)((acc[f][r] + bb) * scale);
        }
    }
}

// ---------------------------------------------------------------------------
// Kernel 4: windowed attention. One wave (32 threads) per (b, h, window, query-slab).
// grid = B*H*NW*4 = 8192 blocks. Each wave owns 16 queries of one window.
// Scores (<=576 cols) live in static LDS; softmax scalar; PV via WMMA.
// ---------------------------------------------------------------------------
__global__ __launch_bounds__(32) void attn_kernel(const __bf16* __restrict__ qkv,
                                                  const float* __restrict__ kpm,
                                                  float* __restrict__ out) {
    const __bf16* qp = qkv;
    const __bf16* kp = qkv + (size_t)Mtot * Dsz;
    const __bf16* vp = qkv + (size_t)2 * Mtot * Dsz;

    const int x  = blockIdx.x;
    const int wq = x & 3;            // query block (16 rows) within window
    const int w  = (x >> 2) & 31;    // window index
    const int h  = (x >> 7) & 15;    // head
    const int b  = x >> 11;          // batch

    const int lane = threadIdx.x & 31;
    const int hi   = lane >> 4;
    const int nn   = lane & 15;

    const size_t bh = (size_t)b * Hsz + h;
    const __bf16* qbase = qp + bh * Ssz * HD;
    const __bf16* kbase = kp + bh * Ssz * HD;
    const __bf16* vbase = vp + bh * Ssz * HD;

    __shared__ float sm[16][580];    // 16 query rows x up to 576 score cols

    // --- Q fragments for this wave's 16 rows, K-dim = hd split into 2x32 ---
    const int srow = w * WIN + wq * 16 + nn;
    BF16x16 a0, a1;
    {
        const __bf16* p0 = qbase + (size_t)srow * HD + hi * 8;
        a0.q[0] = *(const uint4*)(p0);
        a0.q[1] = *(const uint4*)(p0 + 16);
        const __bf16* p1 = p0 + 32;
        a1.q[0] = *(const uint4*)(p1);
        a1.q[1] = *(const uint4*)(p1 + 16);
    }

    const int ncols = WIN + w * 16;  // 64 local + 16 per earlier window

    // --- local score tiles: S = Q K^T (scale already folded into Q) ---
    for (int kt = 0; kt < 4; ++kt) {
        const int key = w * WIN + kt * 16 + nn;
        BF16x16 b0, b1;
        const __bf16* kb0 = kbase + (size_t)key * HD + hi * 8;
        b0.q[0] = *(const uint4*)(kb0);
        b0.q[1] = *(const uint4*)(kb0 + 16);
        const __bf16* kb1 = kb0 + 32;
        b1.q[0] = *(const uint4*)(kb1);
        b1.q[1] = *(const uint4*)(kb1 + 16);

        f32x8 acc = {};
        acc = WMMA_BF16(a0.v, b0.v, acc);
        acc = WMMA_BF16(a1.v, b1.v, acc);

        const float kpv     = kpm[(size_t)b * Ssz + key];
        const bool  allowed = (wq >= kt);   // block-causal: query block >= key block
#pragma unroll
        for (int r = 0; r < 8; ++r) {
            const int M = r + hi * 8;
            sm[M][kt * 16 + nn] = allowed ? (acc[r] + kpv) : NEGV;
        }
    }

    // --- global score tiles: last 16 keys of every earlier window (gwin < w) ---
    for (int gt = 0; gt < w; ++gt) {
        const int key = gt * WIN + 48 + nn;
        if (gt + 1 < w)  // hint the next strided K tile into cache
            __builtin_prefetch(kbase + (size_t)((gt + 1) * WIN + 48 + nn) * HD, 0, 0);

        BF16x16 b0, b1;
        const __bf16* kb0 = kbase + (size_t)key * HD + hi * 8;
        b0.q[0] = *(const uint4*)(kb0);
        b0.q[1] = *(const uint4*)(kb0 + 16);
        const __bf16* kb1 = kb0 + 32;
        b1.q[0] = *(const uint4*)(kb1);
        b1.q[1] = *(const uint4*)(kb1 + 16);

        f32x8 acc = {};
        acc = WMMA_BF16(a0.v, b0.v, acc);
        acc = WMMA_BF16(a1.v, b1.v, acc);

        const float kpv = kpm[(size_t)b * Ssz + key];
#pragma unroll
        for (int r = 0; r < 8; ++r) {
            const int M = r + hi * 8;
            sm[M][64 + gt * 16 + nn] = acc[r] + kpv;
        }
    }
    __syncthreads();

    // --- softmax: lanes 0..15 each own one query row ---
    if (lane < 16) {
        const int m = lane;
        float mx = -3.0e38f;
        for (int c = 0; c < ncols; ++c) mx = fmaxf(mx, sm[m][c]);
        float sum = 0.0f;
        for (int c = 0; c < ncols; ++c) {
            const float e = __expf(sm[m][c] - mx);
            sm[m][c] = e;
            sum += e;
        }
        const float inv = 1.0f / sum;
        for (int c = 0; c < ncols; ++c) sm[m][c] *= inv;
    }
    __syncthreads();

    // --- PV: out(16 x 64) = P(16 x ncols) @ Vcat(ncols x 64), WMMA over key tiles ---
    f32x8 oacc[4] = {{}, {}, {}, {}};
    for (int kk = 0; kk < ncols; kk += 32) {
        BF16x16 pa;
        const int m = nn;   // A-matrix: M = lane % 16 for both lane halves
#pragma unroll
        for (int i = 0; i < 16; ++i) {
            const int kel = kk + ((i < 8) ? i : i + 8) + hi * 8;
            pa.e[i] = (kel < ncols) ? (__bf16)sm[m][kel] : (__bf16)0.0f;
        }
#pragma unroll
        for (int nt = 0; nt < 4; ++nt) {
            BF16x16 vb;
            const int d = nt * 16 + nn;
#pragma unroll
            for (int i = 0; i < 16; ++i) {
                const int kel = kk + ((i < 8) ? i : i + 8) + hi * 8;
                int s;
                if (kel < WIN) {
                    s = w * WIN + kel;                       // local key
                } else {
                    const int g = kel - WIN;                 // global key
                    s = (g >> 4) * WIN + 48 + (g & 15);
                }
                const bool ok = (kel < ncols);
                s = ok ? s : 0;
                const __bf16 val = vbase[(size_t)s * HD + d];
                vb.e[i] = ok ? val : (__bf16)0.0f;
            }
            oacc[nt] = WMMA_BF16(pa.v, vb.v, oacc[nt]);
        }
    }

    // --- store fp32 output (B, S, H*hd) ---
#pragma unroll
    for (int nt = 0; nt < 4; ++nt) {
        const int d = nt * 16 + nn;
#pragma unroll
        for (int r = 0; r < 8; ++r) {
            const int M  = r + hi * 8;
            const int sq = w * WIN + wq * 16 + M;
            out[((size_t)b * Ssz + sq) * Dsz + h * HD + d] = oacc[nt][r];
        }
    }
}

// ---------------------------------------------------------------------------
// Host launcher
// ---------------------------------------------------------------------------
extern "C" void kernel_launch(void* const* d_in, const int* in_sizes, int n_in,
                              void* d_out, int out_size, void* d_ws, size_t ws_size,
                              hipStream_t stream) {
    (void)in_sizes; (void)n_in; (void)out_size; (void)ws_size;

    const float* x   = (const float*)d_in[0];
    const float* kpm = (const float*)d_in[1];
    const float* Wq  = (const float*)d_in[2];
    const float* bq  = (const float*)d_in[3];
    const float* Wk  = (const float*)d_in[4];
    const float* bk  = (const float*)d_in[5];
    const float* Wv  = (const float*)d_in[6];
    const float* bv  = (const float*)d_in[7];
    float* out = (float*)d_out;

    char* ws = (char*)d_ws;
    __bf16* xb  = (__bf16*)ws;                                   // 16 MB: x in bf16
    __bf16* wt  = (__bf16*)(ws + (size_t)16 * 1024 * 1024);      //  6 MB: Wt q/k/v bf16
    __bf16* qkv = (__bf16*)(ws + (size_t)24 * 1024 * 1024);      // 48 MB: Q,K,V (B,H,S,hd) bf16

    const int nX = Mtot * Dsz;                                   // 8,388,608
    cvt_x_kernel<<<(nX + 255) / 256, 256, 0, stream>>>(x, xb, nX);

    cvt_wt_kernel<<<dim3((Dsz * Dsz) / 256, 1, 3), 256, 0, stream>>>(Wq, Wk, Wv, wt);

    proj_kernel<<<dim3(Mtot / 128, Dsz / 64, 3), 256, 0, stream>>>(xb, wt, bq, bk, bv, qkv);

    attn_kernel<<<Bsz * Hsz * NW * 4, 32, 0, stream>>>(qkv, kpm, out);
}
